// ExternalAttention_60163901882779
// MI455X (gfx1250) — matrix operand order repaired and checked by reading, not compiled, not run on hardware
//
#include <hip/hip_runtime.h>

// ---------------------------------------------------------------------------
// ExternalAttention for MI455X (gfx1250, wave32, WMMA).
//   D_MODEL=1024, H=16, D_HEAD=64, MEM=256, B=4, T=4096.
// All heavy math runs on v_wmma_f32_16x16x32_bf16. The two K=1024 projections
// use a hi/lo split-bf16 3-product scheme for ~fp32 accuracy.
// ---------------------------------------------------------------------------

typedef __attribute__((ext_vector_type(16))) __bf16 v16bf;
typedef __attribute__((ext_vector_type(8)))  __bf16 bf16x8;
typedef __attribute__((ext_vector_type(8)))  float  v8f;

#define BATCH   4
#define T_LEN   4096
#define D_MODEL 1024
#define N_HEADS 16
#define D_HEAD  64
#define MEM     256
#define BT      (BATCH * T_LEN)          // 16384 rows

__device__ __forceinline__ v8f wmma_bf16(v16bf a, v16bf b, v8f c) {
  // (neg_a, A, neg_b, B, c_mod, C, reuse_a, reuse_b)
  return __builtin_amdgcn_wmma_f32_16x16x32_bf16(false, a, false, b, (short)0, c,
                                                 false, false);
}

// A fragment: 16x32 bf16, rows M contiguous-K in memory (lda elements).
// Lane L(0-15): row M=L, K = [k0..k0+15]
// Lane L+16   : row M=L, K = [k0+16..k0+31]
__device__ __forceinline__ v16bf load_a_frag(const __bf16* A, int lda,
                                             int row0, int k0, int lane) {
  int r  = row0 + (lane & 15);
  int kk = k0 + ((lane >> 4) << 4);
  const __bf16* p = A + (size_t)r * lda + kk;
  union { bf16x8 h2[2]; v16bf v; } u;
  u.h2[0] = *(const bf16x8*)(p);
  u.h2[1] = *(const bf16x8*)(p + 8);
  return u.v;
}

// B fragment: 32x16 bf16 (KxN), B stored as rows of N each K-contiguous (ldb).
// Lane L(0-15): col N=L, K=[k0..k0+15];  Lane L+16: col N=L, K=[k0+16..k0+31]
__device__ __forceinline__ v16bf load_b_frag(const __bf16* B, int ldb,
                                             int col0, int k0, int lane) {
  int c  = col0 + (lane & 15);
  int kk = k0 + ((lane >> 4) << 4);
  return *(const v16bf*)(B + (size_t)c * ldb + kk);
}

// ---------------------------------------------------------------------------
// Kernel 0a: split fp32 -> bf16 hi + bf16 lo (residual)
// ---------------------------------------------------------------------------
__global__ void split_bf16_kernel(const float* __restrict__ src,
                                  __bf16* __restrict__ hi,
                                  __bf16* __restrict__ lo, int n) {
  for (int i = blockIdx.x * blockDim.x + threadIdx.x; i < n;
       i += gridDim.x * blockDim.x) {
    float v = src[i];
    __bf16 h = (__bf16)v;
    hi[i] = h;
    lo[i] = (__bf16)(v - (float)h);
  }
}

// ---------------------------------------------------------------------------
// Kernel 0b: M_k -> bf16 (same layout), M_v -> bf16 transposed per head:
//   mvt[h][d][m] = M_v[h][m][d]   (so the output GEMM's B is K-contiguous)
// ---------------------------------------------------------------------------
__global__ void conv_mem_kernel(const float* __restrict__ Mk,
                                const float* __restrict__ Mv,
                                __bf16* __restrict__ mkb,
                                __bf16* __restrict__ mvt) {
  int i = blockIdx.x * blockDim.x + threadIdx.x;   // 0 .. 16*256*64-1
  mkb[i] = (__bf16)Mk[i];
  int h = i >> 14;           // head
  int m = (i >> 6) & (MEM - 1);
  int d = i & (D_HEAD - 1);
  mvt[((size_t)h << 14) + (d << 8) + m] = (__bf16)Mv[i];
}

// ---------------------------------------------------------------------------
// Kernel 1/4: C[M,N] = A[M,K] * B[N,K]^T with split-bf16 x3 compensation.
// One 16x16 output tile per wave; the 192MB L2 provides operand reuse.
// ---------------------------------------------------------------------------
__global__ void gemm_bf16x3_kernel(const __bf16* __restrict__ ah,
                                   const __bf16* __restrict__ al,
                                   const __bf16* __restrict__ bh,
                                   const __bf16* __restrict__ bl,
                                   int M, int N, int K,
                                   float* __restrict__ outF,
                                   __bf16* __restrict__ outBf) {
  int lane = threadIdx.x & 31;
  int wave = blockIdx.x * (blockDim.x >> 5) + (threadIdx.x >> 5);
  int ntn  = N >> 4;
  int tm   = (wave / ntn) << 4;
  int tn   = (wave % ntn) << 4;
  if (tm >= M) return;

  v8f acc = {};
  for (int k = 0; k < K; k += 32) {
    v16bf Ah = load_a_frag(ah, K, tm, k, lane);
    v16bf Al = load_a_frag(al, K, tm, k, lane);
    v16bf Bh = load_b_frag(bh, K, tn, k, lane);
    v16bf Bl = load_b_frag(bl, K, tn, k, lane);
    acc = wmma_bf16(Ah, Bh, acc);     // hi*hi
    acc = wmma_bf16(Ah, Bl, acc);     // hi*lo
    acc = wmma_bf16(Al, Bh, acc);     // lo*hi   (lo*lo term ~2^-18: dropped)
  }

  // C/D layout: lanes 0-15 hold N=lane rows M=tm+r; lanes 16-31 rows M=tm+8+r
  int c  = tn + (lane & 15);
  int rb = tm + ((lane >> 4) << 3);
#pragma unroll
  for (int r = 0; r < 8; ++r) {
    float v = acc[r];
    size_t idx = (size_t)(rb + r) * N + c;
    if (outF)  outF[idx]  = v;
    if (outBf) outBf[idx] = (__bf16)v;
  }
}

// ---------------------------------------------------------------------------
// Shared softmax-tile computation: for a wave's (b,h,16-token tile) compute
// w[j][r] = softmax over m of (Q . M_k / 8), held in C-layout registers:
//   lane<16 : rows t0..t0+7,  column m = 16*j + lane
//   lane>=16: rows t0+8..t0+15, same column
// ---------------------------------------------------------------------------
__device__ __forceinline__ void score_softmax_tile(const __bf16* __restrict__ Qb,
                                                   const __bf16* __restrict__ Kh,
                                                   int b, int h, int t0, int lane,
                                                   float w[16][8]) {
  const __bf16* Qrow = Qb + (size_t)b * T_LEN * D_MODEL + h * D_HEAD;
  v16bf a0 = load_a_frag(Qrow, D_MODEL, t0, 0, lane);
  v16bf a1 = load_a_frag(Qrow, D_MODEL, t0, 32, lane);

#pragma unroll
  for (int j = 0; j < 16; ++j) {
    v8f acc = {};
    v16bf b0 = load_b_frag(Kh, D_HEAD, j << 4, 0, lane);
    v16bf b1 = load_b_frag(Kh, D_HEAD, j << 4, 32, lane);
    acc = wmma_bf16(a0, b0, acc);
    acc = wmma_bf16(a1, b1, acc);
#pragma unroll
    for (int r = 0; r < 8; ++r) w[j][r] = acc[r] * 0.125f;   // D_HEAD^-0.5
  }

  // Row-wise softmax over m (16 lanes of a half x 16 j-tiles per row).
#pragma unroll
  for (int r = 0; r < 8; ++r) {
    float mx = -3.4e38f;
#pragma unroll
    for (int j = 0; j < 16; ++j) mx = fmaxf(mx, w[j][r]);
    for (int off = 8; off >= 1; off >>= 1)
      mx = fmaxf(mx, __shfl_xor(mx, off, 32));   // reduce within 16-lane half
    float s = 0.f;
#pragma unroll
    for (int j = 0; j < 16; ++j) { w[j][r] = __expf(w[j][r] - mx); s += w[j][r]; }
    for (int off = 8; off >= 1; off >>= 1) s += __shfl_xor(s, off, 32);
    float inv = 1.f / s;
#pragma unroll
    for (int j = 0; j < 16; ++j) w[j][r] *= inv;
  }
}

// ---------------------------------------------------------------------------
// Kernel 2: pass 1 — softmax weights, deterministic per-tile column sums.
// partial[bh][ttile][m]   (reduced in fixed order by kernel 3)
// ---------------------------------------------------------------------------
__global__ void attn_pass1_kernel(const __bf16* __restrict__ Qb,
                                  const __bf16* __restrict__ mkb,
                                  float* __restrict__ partial) {
  int lane = threadIdx.x & 31;
  int wave = blockIdx.x * (blockDim.x >> 5) + (threadIdx.x >> 5);
  int tt   = wave & (T_LEN / 16 - 1);
  int bh   = wave / (T_LEN / 16);
  int h    = bh & (N_HEADS - 1);
  int b    = bh >> 4;
  int t0   = tt << 4;
  const __bf16* Kh = mkb + (size_t)h * MEM * D_HEAD;

  float w[16][8];
  score_softmax_tile(Qb, Kh, b, h, t0, lane, w);

#pragma unroll
  for (int j = 0; j < 16; ++j) {
    float cs = 0.f;
#pragma unroll
    for (int r = 0; r < 8; ++r) cs += w[j][r];
    cs += __shfl_xor(cs, 16, 32);                // combine both row-halves
    if (lane < 16)
      partial[((size_t)bh * (T_LEN / 16) + tt) * MEM + (j << 4) + lane] = cs;
  }
}

// ---------------------------------------------------------------------------
// Kernel 3: fixed-order reduction over token tiles -> colsum[bh][m]
// ---------------------------------------------------------------------------
__global__ void colsum_reduce_kernel(const float* __restrict__ partial,
                                     float* __restrict__ colsum) {
  int i  = blockIdx.x * blockDim.x + threadIdx.x;  // 0..BATCH*N_HEADS*MEM-1
  int bh = i >> 8;
  int m  = i & (MEM - 1);
  const float* p = partial + (size_t)bh * (T_LEN / 16) * MEM + m;
  float s = 0.f;
  for (int t = 0; t < T_LEN / 16; ++t) s += p[(size_t)t * MEM];
  colsum[i] = s;
}

// ---------------------------------------------------------------------------
// Kernel 4: pass 2 — recompute softmax (bit-identical), L1-normalize with
// colsum, then out_tile = w_norm * M_v via WMMA (A staged through LDS to
// convert C-layout -> A-layout). Emits attn as split bf16 hi/lo for GEMM-2.
// 4 waves / block -> 32 KB static LDS.
// ---------------------------------------------------------------------------
__global__ void attn_pass2_kernel(const __bf16* __restrict__ Qb,
                                  const __bf16* __restrict__ mkb,
                                  const __bf16* __restrict__ mvt,
                                  const float* __restrict__ colsum,
                                  __bf16* __restrict__ attnh,
                                  __bf16* __restrict__ attnl) {
  __shared__ __bf16 sW[4][16][MEM];                // 32 KB
  int lane = threadIdx.x & 31;
  int widx = threadIdx.x >> 5;
  int wave = blockIdx.x * (blockDim.x >> 5) + widx;
  int tt   = wave & (T_LEN / 16 - 1);
  int bh   = wave / (T_LEN / 16);
  int h    = bh & (N_HEADS - 1);
  int b    = bh >> 4;
  int t0   = tt << 4;
  const __bf16* Kh = mkb + (size_t)h * MEM * D_HEAD;

  float w[16][8];
  score_softmax_tile(Qb, Kh, b, h, t0, lane, w);

  // normalize by token-column sums, stash bf16 tile in LDS (row-major 16xMEM)
  int rb = (lane >> 4) << 3;
#pragma unroll
  for (int j = 0; j < 16; ++j) {
    float inv = 1.f / (colsum[(bh << 8) + (j << 4) + (lane & 15)] + 1e-6f);
#pragma unroll
    for (int r = 0; r < 8; ++r)
      sW[widx][rb + r][(j << 4) + (lane & 15)] = (__bf16)(w[j][r] * inv);
  }
  __syncthreads();   // uniform; guarantees DS ordering for cross-lane reads

  // out[16 x 64] = w_norm[16 x 256] * M_v  (B = mvt[h][d][m], K=m contiguous)
  const __bf16* Bv = mvt + ((size_t)h << 14);
  v8f acc[4] = {v8f{}, v8f{}, v8f{}, v8f{}};
#pragma unroll
  for (int mc = 0; mc < 8; ++mc) {                 // K = 256 in chunks of 32
    const __bf16* lp = &sW[widx][lane & 15][(mc << 5) + ((lane >> 4) << 4)];
    union { bf16x8 h2[2]; v16bf v; } u;
    u.h2[0] = *(const bf16x8*)(lp);
    u.h2[1] = *(const bf16x8*)(lp + 8);
#pragma unroll
    for (int dt = 0; dt < 4; ++dt) {
      v16bf Bf = load_b_frag(Bv, MEM, dt << 4, mc << 5, lane);
      acc[dt] = wmma_bf16(u.v, Bf, acc[dt]);
    }
  }

  // store split-bf16 attn: row = b*T + t, col = h*64 + d  (K-contig for GEMM2)
#pragma unroll
  for (int dt = 0; dt < 4; ++dt)
#pragma unroll
    for (int r = 0; r < 8; ++r) {
      float v = acc[dt][r];
      __bf16 hi = (__bf16)v;
      size_t idx = (size_t)(b * T_LEN + t0 + rb + r) * D_MODEL + h * D_HEAD +
                   (dt << 4) + (lane & 15);
      attnh[idx] = hi;
      attnl[idx] = (__bf16)(v - (float)hi);
    }
}

// ---------------------------------------------------------------------------
extern "C" void kernel_launch(void* const* d_in, const int* in_sizes, int n_in,
                              void* d_out, int out_size, void* d_ws, size_t ws_size,
                              hipStream_t stream) {
  (void)in_sizes; (void)n_in; (void)out_size; (void)ws_size;
  const float* x  = (const float*)d_in[0];
  const float* Wq = (const float*)d_in[1];
  const float* Mk = (const float*)d_in[2];
  const float* Mv = (const float*)d_in[3];
  const float* Wo = (const float*)d_in[4];
  float* out = (float*)d_out;

  char* base = (char*)d_ws;
  size_t off = 0;
  auto take = [&](size_t bytes) -> char* {
    char* p = base + off;
    off += (bytes + 255) & ~(size_t)255;
    return p;
  };
  const size_t nX = (size_t)BT * D_MODEL;          // 16.7M
  const size_t nW = (size_t)D_MODEL * D_MODEL;     // 1.05M
  const size_t nM = (size_t)N_HEADS * MEM * D_HEAD;

  __bf16* xh  = (__bf16*)take(nX * 2);   // reused as attn_hi after GEMM-1
  __bf16* xl  = (__bf16*)take(nX * 2);   // reused as attn_lo after GEMM-1
  __bf16* wqh = (__bf16*)take(nW * 2);
  __bf16* wql = (__bf16*)take(nW * 2);
  __bf16* woh = (__bf16*)take(nW * 2);
  __bf16* wol = (__bf16*)take(nW * 2);
  __bf16* mkb = (__bf16*)take(nM * 2);
  __bf16* mvt = (__bf16*)take(nM * 2);
  __bf16* qbf = (__bf16*)take(nX * 2);
  float* partial = (float*)take((size_t)BATCH * N_HEADS * (T_LEN / 16) * MEM * 4);
  float* colsum  = (float*)take((size_t)BATCH * N_HEADS * MEM * 4);

  // 0) precision-split / convert operands
  split_bf16_kernel<<<4096, 256, 0, stream>>>(x,  xh,  xl,  (int)nX);
  split_bf16_kernel<<<1024, 256, 0, stream>>>(Wq, wqh, wql, (int)nW);
  split_bf16_kernel<<<1024, 256, 0, stream>>>(Wo, woh, wol, (int)nW);
  conv_mem_kernel<<<(int)(nM / 256), 256, 0, stream>>>(Mk, Mv, mkb, mvt);

  // 1) Q = x @ Wq^T  (split-bf16 x3; bf16 output for attention)
  gemm_bf16x3_kernel<<<(BT / 16) * (D_MODEL / 16) / 8, 256, 0, stream>>>(
      xh, xl, wqh, wql, BT, D_MODEL, D_MODEL, nullptr, qbf);

  // 2) softmax weights + deterministic column partial sums
  attn_pass1_kernel<<<BATCH * N_HEADS * (T_LEN / 16) / 8, 256, 0, stream>>>(
      qbf, mkb, partial);

  // 3) colsum[bh][m] = sum_t w  (fixed order)
  colsum_reduce_kernel<<<BATCH * N_HEADS * MEM / 256, 256, 0, stream>>>(
      partial, colsum);

  // 4) renormalize + w @ M_v  -> attn (split bf16), reusing xh/xl storage
  attn_pass2_kernel<<<BATCH * N_HEADS * (T_LEN / 16) / 4, 128, 0, stream>>>(
      qbf, mkb, mvt, colsum, xh, xl);

  // 5) out = attn @ Wo^T  (split-bf16 x3, fp32 result)
  gemm_bf16x3_kernel<<<(BT / 16) * (D_MODEL / 16) / 8, 256, 0, stream>>>(
      xh, xl, woh, wol, BT, D_MODEL, D_MODEL, out, nullptr);
}
